// AdvancedRecipeGenerator_28724741276231
// MI455X (gfx1250) — compile-verified, matrix operand order checked by reading, not run on hardware
//
#include <hip/hip_runtime.h>
#include <hip/hip_bf16.h>
#include <math.h>

// ---------------------------------------------------------------------------
// Types for CDNA5 WMMA
// ---------------------------------------------------------------------------
typedef __attribute__((ext_vector_type(16))) __bf16 v16bf;
typedef __attribute__((ext_vector_type(8)))  float  v8f;

union Frag16 {
    v16bf    v;
    unsigned u[8];
};

__device__ __forceinline__ unsigned bf16_of(float x) {
    unsigned u = __float_as_uint(x);
    unsigned r = (u + 0x7FFFu + ((u >> 16) & 1u)) >> 16;   // round-nearest-even
    return r & 0xFFFFu;
}
__device__ __forceinline__ unsigned pack_bf16(float lo, float hi) {
    return bf16_of(lo) | (bf16_of(hi) << 16);
}
__device__ __forceinline__ float gelu_exact(float x) {
    return 0.5f * x * (1.0f + erff(x * 0.70710678118654752f));
}

// epilogue flags
#define EPI_BIAS 1
#define EPI_GELU 2

// ---------------------------------------------------------------------------
// Generic GEMM: C[M,N] = epi(A[M,K] @ W[K,N] + bias)
// Requirements: M % 64 == 0, K % 32 == 0. N arbitrary (guarded).
// bf16 staging in LDS, f32 accumulate via v_wmma_f32_16x16x32_bf16.
// Batched over blockIdx.z with element strides.
// ---------------------------------------------------------------------------
#define LSTR 18   // uints (bf16 pairs) per LDS row: 16 pairs + 2 pad

__global__ __launch_bounds__(128)
void gemm_bf16_wmma(const float* __restrict__ A, const float* __restrict__ W,
                    const float* __restrict__ bias, float* __restrict__ C,
                    int M, int N, int K, int flags,
                    long aStride, long wStride, long bStride, long cStride)
{
    __shared__ unsigned sA[64 * LSTR];
    __shared__ unsigned sB[64 * LSTR];

    const int z = blockIdx.z;
    A += (long)z * aStride;
    W += (long)z * wStride;
    C += (long)z * cStride;
    if (flags & EPI_BIAS) bias += (long)z * bStride;

    const int tid  = threadIdx.x;
    const int lane = tid & 31;
    const int wave = tid >> 5;
    const int wm   = wave >> 1;          // 2x2 wave grid over 64x64 tile
    const int wn   = wave & 1;
    const int bm0  = blockIdx.y * 64;
    const int bn0  = blockIdx.x * 64;
    const int lr   = lane & 15;
    const int hi   = (lane >> 4) & 1;

    v8f acc[2][2] = {};

    for (int kt = 0; kt < K; kt += 32) {
        __syncthreads();
        // ---- stage A tile [64 rows x 32 k] as bf16 pairs (row-major, K-major pairs)
        for (int idx = tid; idx < 64 * 16; idx += 128) {
            int r = idx >> 4, p = idx & 15;
            const float2 f = *(const float2*)(A + (long)(bm0 + r) * K + kt + 2 * p);
            sA[r * LSTR + p] = pack_bf16(f.x, f.y);
        }
        // ---- stage B tile [32 k x 64 n] transposed into N-major / K-pair layout
        for (int idx = tid; idx < 64 * 16; idx += 128) {
            int p  = idx >> 6;           // 0..15 k-pair
            int nn = idx & 63;           // 0..63 column (coalesced global read)
            int gn = bn0 + nn;
            float lo = 0.f, hh = 0.f;
            if (gn < N) {
                lo = W[(long)(kt + 2 * p)     * N + gn];
                hh = W[(long)(kt + 2 * p + 1) * N + gn];
            }
            sB[nn * LSTR + p] = pack_bf16(lo, hh);
        }
        if (kt + 32 < K) {
            __builtin_prefetch(A + (long)(bm0 + (tid & 63)) * K + kt + 32, 0, 1);
        }
        __syncthreads();

        // ---- gather fragments (ISA 16-bit A/B layout, wave32)
        Frag16 af[2], bf[2];
        #pragma unroll
        for (int t = 0; t < 2; ++t) {
            const int mrow = wm * 32 + t * 16 + lr;
            const int nrow = wn * 32 + t * 16 + lr;
            #pragma unroll
            for (int v = 0; v < 8; ++v) {
                const int p = (v & 3) + ((v >> 2) << 3) + (hi << 2);
                af[t].u[v] = sA[mrow * LSTR + p];
                bf[t].u[v] = sB[nrow * LSTR + p];
            }
        }
        #pragma unroll
        for (int ti = 0; ti < 2; ++ti)
            #pragma unroll
            for (int tj = 0; tj < 2; ++tj)
                acc[ti][tj] = __builtin_amdgcn_wmma_f32_16x16x32_bf16(
                    false, af[ti].v, false, bf[tj].v,
                    (short)0, acc[ti][tj], false, false);
    }

    // ---- epilogue: C f32 16x16 layout => row = r + 8*hi, col = lane&15
    #pragma unroll
    for (int ti = 0; ti < 2; ++ti) {
        #pragma unroll
        for (int tj = 0; tj < 2; ++tj) {
            const int col = bn0 + wn * 32 + tj * 16 + lr;
            if (col >= N) continue;
            const float bv = (flags & EPI_BIAS) ? bias[col] : 0.0f;
            #pragma unroll
            for (int r = 0; r < 8; ++r) {
                const int row = bm0 + wm * 32 + ti * 16 + r + 8 * hi;
                float v = acc[ti][tj][r] + bv;
                if (flags & EPI_GELU) v = gelu_exact(v);
                C[(long)row * N + col] = v;
            }
        }
    }
}

// ---------------------------------------------------------------------------
// Build x[8192,512]: token0 = text_f + PE(0), token1 = emo_emb[argmax] + PE(1)
// ---------------------------------------------------------------------------
__global__ __launch_bounds__(256)
void build_x_kernel(const float* __restrict__ text_f,
                    const float* __restrict__ emo_scores,
                    const float* __restrict__ emo_emb,
                    float* __restrict__ x)
{
    const int b = blockIdx.x;
    __shared__ int sidx;
    if (threadIdx.x == 0) {
        int best = 0;
        float bv = emo_scores[b * 10];
        for (int e = 1; e < 10; ++e) {
            float v = emo_scores[b * 10 + e];
            if (v > bv) { bv = v; best = e; }
        }
        sidx = best;
    }
    __syncthreads();
    const int idx = sidx;
    for (int d = threadIdx.x; d < 512; d += 256) {
        const int   i   = d >> 1;
        const float ang = expf(-logf(10000.0f) * (2.0f * (float)i) / 512.0f); // pos=1 angle
        const float pe0 = (d & 1) ? 1.0f : 0.0f;                  // sin(0)=0, cos(0)=1
        const float pe1 = (d & 1) ? cosf(ang) : sinf(ang);
        x[((long)(2 * b))     * 512 + d] = text_f[(long)b * 512 + d]    + pe0;
        x[((long)(2 * b) + 1) * 512 + d] = emo_emb[(long)idx * 512 + d] + pe1;
    }
}

// ---------------------------------------------------------------------------
// LayerNorm over D=512 (optionally fused residual add): dst = LN(src (+ res))
// ---------------------------------------------------------------------------
__global__ __launch_bounds__(128)
void layernorm_kernel(const float* __restrict__ src, const float* __restrict__ res,
                      const float* __restrict__ g, const float* __restrict__ b,
                      float* __restrict__ dst)
{
    const int row = blockIdx.x, tid = threadIdx.x;
    __shared__ float s1[128], s2[128];
    const float* sp = src + (long)row * 512;
    const float* rp = res ? res + (long)row * 512 : nullptr;
    float v[4]; float sum = 0.f, sq = 0.f;
    #pragma unroll
    for (int i = 0; i < 4; ++i) {
        const int d = tid + i * 128;
        float t = sp[d] + (rp ? rp[d] : 0.0f);
        v[i] = t; sum += t; sq += t * t;
    }
    s1[tid] = sum; s2[tid] = sq; __syncthreads();
    for (int s = 64; s > 0; s >>= 1) {
        if (tid < s) { s1[tid] += s1[tid + s]; s2[tid] += s2[tid + s]; }
        __syncthreads();
    }
    const float mu   = s1[0] * (1.0f / 512.0f);
    const float var  = s2[0] * (1.0f / 512.0f) - mu * mu;
    const float rstd = rsqrtf(var + 1e-5f);
    #pragma unroll
    for (int i = 0; i < 4; ++i) {
        const int d = tid + i * 128;
        dst[(long)row * 512 + d] = (v[i] - mu) * rstd * g[d] + b[d];
    }
}

// ---------------------------------------------------------------------------
// MoE: per-expert LN of y[t] then out[t] += combine[t,e] * LN(y[t])
// ---------------------------------------------------------------------------
__global__ __launch_bounds__(128)
void moe_ln_acc_kernel(const float* __restrict__ y, const float* __restrict__ g,
                       const float* __restrict__ b, const float* __restrict__ comb,
                       int e, float* __restrict__ out)
{
    const int row = blockIdx.x, tid = threadIdx.x;
    const float w = comb[(long)row * 8 + e];
    __shared__ float s1[128], s2[128];
    const float* sp = y + (long)row * 512;
    float v[4]; float sum = 0.f, sq = 0.f;
    #pragma unroll
    for (int i = 0; i < 4; ++i) {
        const int d = tid + i * 128;
        float t = sp[d];
        v[i] = t; sum += t; sq += t * t;
    }
    s1[tid] = sum; s2[tid] = sq; __syncthreads();
    for (int s = 64; s > 0; s >>= 1) {
        if (tid < s) { s1[tid] += s1[tid + s]; s2[tid] += s2[tid + s]; }
        __syncthreads();
    }
    const float mu   = s1[0] * (1.0f / 512.0f);
    const float var  = s2[0] * (1.0f / 512.0f) - mu * mu;
    const float rstd = rsqrtf(var + 1e-5f);
    #pragma unroll
    for (int i = 0; i < 4; ++i) {
        const int d = tid + i * 128;
        out[(long)row * 512 + d] += w * ((v[i] - mu) * rstd * g[d] + b[d]);
    }
}

// ---------------------------------------------------------------------------
// MHA core for tiny sequences (Sq,Sk <= 2), H=8, DK=64, D=512.
// Q:[B,Sq,512], K/V:[B,Sk,512] -> O:[B,Sq,512]. grid = B*8, block = 64.
// ---------------------------------------------------------------------------
__global__ __launch_bounds__(64)
void attn_kernel(const float* __restrict__ Q, const float* __restrict__ K,
                 const float* __restrict__ V, float* __restrict__ O,
                 int Sq, int Sk)
{
    const int bh = blockIdx.x, b = bh >> 3, h = bh & 7, d = threadIdx.x;
    __shared__ float red[64];
    __shared__ float sc[2][2];
    float qv[2], kv[2], vv[2];
    for (int i = 0; i < Sq; ++i) qv[i] = Q[((long)(b * Sq + i)) * 512 + h * 64 + d];
    for (int j = 0; j < Sk; ++j) {
        kv[j] = K[((long)(b * Sk + j)) * 512 + h * 64 + d];
        vv[j] = V[((long)(b * Sk + j)) * 512 + h * 64 + d];
    }
    for (int i = 0; i < Sq; ++i)
        for (int j = 0; j < Sk; ++j) {
            red[d] = qv[i] * kv[j];
            __syncthreads();
            for (int s = 32; s > 0; s >>= 1) {
                if (d < s) red[d] += red[d + s];
                __syncthreads();
            }
            if (d == 0) sc[i][j] = red[0] * 0.125f;   // 1/sqrt(64)
            __syncthreads();
        }
    float attn[2][2];
    for (int i = 0; i < Sq; ++i) {
        float mx = sc[i][0];
        for (int j = 1; j < Sk; ++j) mx = fmaxf(mx, sc[i][j]);
        float se = 0.f;
        for (int j = 0; j < Sk; ++j) { attn[i][j] = expf(sc[i][j] - mx); se += attn[i][j]; }
        for (int j = 0; j < Sk; ++j) attn[i][j] /= se;
    }
    for (int i = 0; i < Sq; ++i) {
        float o = 0.f;
        for (int j = 0; j < Sk; ++j) o += attn[i][j] * vv[j];
        O[((long)(b * Sq + i)) * 512 + h * 64 + d] = o;
    }
}

// ---------------------------------------------------------------------------
// MoE gate: scores = x @ gate_w + gate_b; top-2; softmax(top2) -> dense combine
// ---------------------------------------------------------------------------
__global__ __launch_bounds__(64)
void gate_topk_kernel(const float* __restrict__ x, const float* __restrict__ gw,
                      const float* __restrict__ gb, float* __restrict__ comb)
{
    const int t = blockIdx.x, tid = threadIdx.x;
    __shared__ float red[8 * 64];
    float acc[8] = {};
    for (int k = tid; k < 512; k += 64) {
        const float xv = x[(long)t * 512 + k];
        #pragma unroll
        for (int e = 0; e < 8; ++e) acc[e] += xv * gw[k * 8 + e];
    }
    #pragma unroll
    for (int e = 0; e < 8; ++e) red[e * 64 + tid] = acc[e];
    __syncthreads();
    for (int s = 32; s > 0; s >>= 1) {
        if (tid < s)
            for (int e = 0; e < 8; ++e) red[e * 64 + tid] += red[e * 64 + tid + s];
        __syncthreads();
    }
    if (tid == 0) {
        float sc[8];
        for (int e = 0; e < 8; ++e) sc[e] = red[e * 64] + gb[e];
        int i0 = 0;
        for (int e = 1; e < 8; ++e) if (sc[e] > sc[i0]) i0 = e;        // first max
        int i1 = -1;
        for (int e = 0; e < 8; ++e)
            if (e != i0 && (i1 < 0 || sc[e] > sc[i1])) i1 = e;          // second
        const float m  = fmaxf(sc[i0], sc[i1]);
        const float e0 = expf(sc[i0] - m), e1 = expf(sc[i1] - m);
        const float s  = e0 + e1;
        for (int e = 0; e < 8; ++e) comb[(long)t * 8 + e] = 0.0f;
        comb[(long)t * 8 + i0] = e0 / s;
        comb[(long)t * 8 + i1] = e1 / s;
    }
}

// ---------------------------------------------------------------------------
// pooled[b] = mean over 2 tokens of enh
// ---------------------------------------------------------------------------
__global__ __launch_bounds__(256)
void pool_kernel(const float* __restrict__ enh, float* __restrict__ pooled)
{
    const int b = blockIdx.x;
    for (int d = threadIdx.x; d < 512; d += 256)
        pooled[(long)b * 512 + d] =
            0.5f * (enh[((long)(2 * b)) * 512 + d] + enh[((long)(2 * b) + 1) * 512 + d]);
}

// ---------------------------------------------------------------------------
// softmax over each group of 200 of nl[4096,600] -> out (note_probs)
// ---------------------------------------------------------------------------
__global__ __launch_bounds__(64)
void softmax200_kernel(const float* __restrict__ nl, float* __restrict__ out)
{
    const int bg = blockIdx.x, b = bg / 3, g = bg % 3, tid = threadIdx.x;
    const float* src = nl  + (long)b * 600 + g * 200;
    float*       dst = out + (long)b * 600 + g * 200;
    __shared__ float red[64];
    float mx = -1e30f;
    for (int i = tid; i < 200; i += 64) mx = fmaxf(mx, src[i]);
    red[tid] = mx; __syncthreads();
    for (int s = 32; s > 0; s >>= 1) {
        if (tid < s) red[tid] = fmaxf(red[tid], red[tid + s]);
        __syncthreads();
    }
    mx = red[0]; __syncthreads();
    float se = 0.f;
    for (int i = tid; i < 200; i += 64) se += expf(src[i] - mx);
    red[tid] = se; __syncthreads();
    for (int s = 32; s > 0; s >>= 1) {
        if (tid < s) red[tid] += red[tid + s];
        __syncthreads();
    }
    const float tot = red[0];
    for (int i = tid; i < 200; i += 64) dst[i] = expf(src[i] - mx) / tot;
}

// ---------------------------------------------------------------------------
// intensity head: out[b] = 10 * sigmoid( h[b,:128] . w2 + b2 )
// ---------------------------------------------------------------------------
__global__ __launch_bounds__(128)
void ip_head_kernel(const float* __restrict__ h, const float* __restrict__ w2,
                    const float* __restrict__ b2, float* __restrict__ out)
{
    const int b = blockIdx.x, tid = threadIdx.x;
    __shared__ float red[128];
    red[tid] = h[(long)b * 128 + tid] * w2[tid];
    __syncthreads();
    for (int s = 64; s > 0; s >>= 1) {
        if (tid < s) red[tid] += red[tid + s];
        __syncthreads();
    }
    if (tid == 0) {
        const float v = red[0] + b2[0];
        out[b] = 10.0f / (1.0f + expf(-v));
    }
}

__global__ void zero_kernel(float* __restrict__ p, long n)
{
    const long i = (long)blockIdx.x * blockDim.x + threadIdx.x;
    if (i < n) p[i] = 0.0f;
}

// ---------------------------------------------------------------------------
// Host orchestration
// ---------------------------------------------------------------------------
static inline void gemm(hipStream_t s, const float* A, const float* W,
                        const float* bias, float* C, int M, int N, int K,
                        int flags, int batch = 1,
                        long aS = 0, long wS = 0, long bS = 0, long cS = 0)
{
    dim3 grid((N + 63) / 64, M / 64, batch);
    gemm_bf16_wmma<<<grid, dim3(128), 0, s>>>(A, W, bias, C, M, N, K, flags,
                                              aS, wS, bS, cS);
}

extern "C" void kernel_launch(void* const* d_in, const int* in_sizes, int n_in,
                              void* d_out, int out_size, void* d_ws, size_t ws_size,
                              hipStream_t stream)
{
    (void)in_sizes; (void)n_in; (void)out_size; (void)ws_size;
    auto P = [&](int i) { return (const float*)d_in[i]; };

    // ---- input/param indices (setup_inputs dict order, depth-first) ----
    const float* text_emb = P(0);            // [4096,768]
    const float* emo_sc   = P(1);            // [4096,10]
    const float* text_w   = P(2);            // [768,512]
    const float* text_b   = P(3);            // [512]
    const float* emo_emb  = P(4);            // [10,512]
    // layers: base 5 + 16*l : wq wk wv wo bq bk bv bo ln1g ln1b ln2g ln2b w1 b1 w2 b2
    const int MOE = 5 + 16 * 6;              // 101
    const float* gate_w = P(MOE + 0);        // [512,8]
    const float* gate_b = P(MOE + 1);        // [8]
    const float* moe_w1 = P(MOE + 2);        // [8,512,1024]
    const float* moe_b1 = P(MOE + 3);        // [8,1024]
    const float* moe_w2 = P(MOE + 4);        // [8,1024,512]
    const float* moe_b2 = P(MOE + 5);        // [8,512]
    const float* moe_lg = P(MOE + 6);        // [8,512]
    const float* moe_lb = P(MOE + 7);        // [8,512]
    const int CR = MOE + 8;                  // cross: wq bq wk bk wv bv wo bo
    const float* lng = P(CR + 8);            // [512]
    const float* lnb = P(CR + 9);
    const int CP = CR + 10;                  // comp: wq bq wk bk wv bv wo bo
    const int HD = CP + 8;
    const float* nc_w1 = P(HD + 0);          // [512,256]
    const float* nc_b1 = P(HD + 1);
    const float* nc_w2 = P(HD + 2);          // [256,600]
    const float* nc_b2 = P(HD + 3);
    const float* ip_w1 = P(HD + 4);          // [512,128]
    const float* ip_b1 = P(HD + 5);
    const float* ip_w2 = P(HD + 6);          // [128,1]
    const float* ip_b2 = P(HD + 7);

    // ---- workspace carving ----
    const long TD = 8192L * 512;             // token-major activation size
    float* ws   = (float*)d_ws;
    size_t off  = 0;
    auto carve  = [&](size_t n) { float* p = ws + off; off += n; return p; };
    float* x    = carve(TD);                 // activations [8192,512]
    float* bufQ = carve(TD);
    float* bufK = carve(TD);
    float* bufV = carve(TD);
    float* bufA = carve(TD);
    float* bufO = carve(TD);
    float* bufF = carve(8192L * 2048);       // FF hidden / reuse region
    float* comb = carve(8192L * 8);          // MoE combine weights
    float* hbuf   = bufF;                    // [8192,1024] expert hidden
    float* crossB = bufF + 8192L * 1024;     // [8192,512]
    float* compB  = crossB;                  // [4096,512] (after cross consumed)
    float* moeO   = bufA;                    // MoE output / pooled reuse

    // ---- 1. embed: text projection + emotion embedding + pos enc ----
    gemm(stream, text_emb, text_w, text_b, bufQ, 4096, 512, 768, EPI_BIAS);
    build_x_kernel<<<4096, 256, 0, stream>>>(bufQ, emo_sc, emo_emb, x);

    // ---- 2. six encoder layers ----
    for (int l = 0; l < 6; ++l) {
        const int Lb = 5 + 16 * l;
        const float *wq = P(Lb+0), *wk = P(Lb+1), *wv = P(Lb+2), *wo = P(Lb+3);
        const float *bq = P(Lb+4), *bk = P(Lb+5), *bv = P(Lb+6), *bo = P(Lb+7);
        const float *g1 = P(Lb+8), *be1 = P(Lb+9), *g2 = P(Lb+10), *be2 = P(Lb+11);
        const float *w1 = P(Lb+12), *b1 = P(Lb+13), *w2 = P(Lb+14), *b2 = P(Lb+15);

        gemm(stream, x, wq, bq, bufQ, 8192, 512, 512, EPI_BIAS);
        gemm(stream, x, wk, bk, bufK, 8192, 512, 512, EPI_BIAS);
        gemm(stream, x, wv, bv, bufV, 8192, 512, 512, EPI_BIAS);
        attn_kernel<<<4096 * 8, 64, 0, stream>>>(bufQ, bufK, bufV, bufA, 2, 2);
        gemm(stream, bufA, wo, bo, bufO, 8192, 512, 512, EPI_BIAS);
        layernorm_kernel<<<8192, 128, 0, stream>>>(bufO, x, g1, be1, x);
        gemm(stream, x, w1, b1, bufF, 8192, 2048, 512, EPI_BIAS | EPI_GELU);
        gemm(stream, bufF, w2, b2, bufO, 8192, 512, 2048, EPI_BIAS);
        layernorm_kernel<<<8192, 128, 0, stream>>>(bufO, x, g2, be2, x);
    }

    // ---- 3. MoE (dense top-2) ----
    gate_topk_kernel<<<8192, 64, 0, stream>>>(x, gate_w, gate_b, comb);
    zero_kernel<<<(int)((TD + 255) / 256), 256, 0, stream>>>(moeO, TD);
    for (int e = 0; e < 8; ++e) {
        gemm(stream, x, moe_w1 + (long)e * 512 * 1024, moe_b1 + (long)e * 1024,
             hbuf, 8192, 1024, 512, EPI_BIAS | EPI_GELU);
        gemm(stream, hbuf, moe_w2 + (long)e * 1024 * 512, moe_b2 + (long)e * 512,
             bufO, 8192, 512, 1024, EPI_BIAS);
        moe_ln_acc_kernel<<<8192, 128, 0, stream>>>(bufO, moe_lg + (long)e * 512,
                                                    moe_lb + (long)e * 512,
                                                    comb, e, moeO);
    }

    // ---- 4. cross attention over moe_out + residual LN -> enh (in x) ----
    gemm(stream, moeO, P(CR+0), P(CR+1), bufQ, 8192, 512, 512, EPI_BIAS);
    gemm(stream, moeO, P(CR+2), P(CR+3), bufK, 8192, 512, 512, EPI_BIAS);
    gemm(stream, moeO, P(CR+4), P(CR+5), bufV, 8192, 512, 512, EPI_BIAS);
    attn_kernel<<<4096 * 8, 64, 0, stream>>>(bufQ, bufK, bufV, bufO, 2, 2);
    gemm(stream, bufO, P(CR+6), P(CR+7), crossB, 8192, 512, 512, EPI_BIAS);
    layernorm_kernel<<<8192, 128, 0, stream>>>(crossB, moeO, lng, lnb, x);

    // ---- 5. pooled + comp attention (Sq=1, Sk=2) ----
    pool_kernel<<<4096, 256, 0, stream>>>(x, moeO);                 // pooled in moeO
    gemm(stream, moeO, P(CP+0), P(CP+1), bufQ, 4096, 512, 512, EPI_BIAS);
    gemm(stream, x,    P(CP+2), P(CP+3), bufK, 8192, 512, 512, EPI_BIAS);
    gemm(stream, x,    P(CP+4), P(CP+5), bufV, 8192, 512, 512, EPI_BIAS);
    attn_kernel<<<4096 * 8, 64, 0, stream>>>(bufQ, bufK, bufV, bufO, 1, 2);
    gemm(stream, bufO, P(CP+6), P(CP+7), compB, 4096, 512, 512, EPI_BIAS);

    // ---- 6. heads ----
    float* out_probs = (float*)d_out;
    float* out_inten = (float*)d_out + 4096L * 600;
    gemm(stream, compB, nc_w1, nc_b1, bufQ, 4096, 256, 512, EPI_BIAS | EPI_GELU);
    gemm(stream, bufQ, nc_w2, nc_b2, bufK, 4096, 600, 256, EPI_BIAS);
    softmax200_kernel<<<4096 * 3, 64, 0, stream>>>(bufK, out_probs);
    gemm(stream, compB, ip_w1, ip_b1, bufV, 4096, 128, 512, EPI_BIAS | EPI_GELU);
    ip_head_kernel<<<4096, 128, 0, stream>>>(bufV, ip_w2, ip_b2, out_inten);
}
